// ScaleGraphBlock_4406636445891
// MI455X (gfx1250) — matrix operand order repaired and checked by reading, not compiled, not run on hardware
//
#include <hip/hip_runtime.h>

// ---------------------------------------------------------------------------
// Types for CDNA5 WMMA (wave32, 16x16x32 f16 -> f32) + TDM descriptors
// ---------------------------------------------------------------------------
typedef _Float16 f16;
typedef __attribute__((ext_vector_type(16))) _Float16 v16h;
typedef __attribute__((ext_vector_type(8)))  _Float16 v8h;
typedef __attribute__((ext_vector_type(8)))  float    v8f;
typedef __attribute__((ext_vector_type(4)))  unsigned int u32x4;
typedef __attribute__((ext_vector_type(4)))  int      i32x4;
typedef __attribute__((ext_vector_type(8)))  int      i32x8;

#if defined(__has_builtin)
#if __has_builtin(__builtin_amdgcn_tensor_load_to_lds) && __has_builtin(__builtin_amdgcn_s_wait_tensorcnt)
#define HAVE_TDM 1
#endif
#endif

#define NB    32          // batch
#define NT    512         // seq len
#define ND    512         // d_model
#define NH    8           // heads
#define NFRM  11          // stft frames = T/HOP + 1
#define NFREQ 257
#define DFTN  514         // 2*257 (re,im interleaved)
#define CHUNK 8192        // stft row chunk
#define NROWS (NB*NT)     // 16384 activation rows
#define GCH   1024        // graph (o,v) channels

__device__ inline float gelu_f(float v) {
  return 0.5f * v * (1.0f + erff(v * 0.70710678118f));
}
__device__ inline v8f zero_v8f() { v8f z; for (int i = 0; i < 8; ++i) z[i] = 0.0f; return z; }
__device__ inline v8h zero_v8h() { v8h z; for (int i = 0; i < 8; ++i) z[i] = (_Float16)0.0f; return z; }

// ---------------------------------------------------------------------------
// Generic WMMA GEMM:  C[M,N] = epi( A[M,K](f16) * B(f16) + bias + resid )
//   blayout=1: B stored [N][K] row-major (x @ W.T pattern, K contiguous)
//   blayout=0: B stored [K][N] row-major
//   epi bits: 1=bias, 2=gelu, 4=residual add
// Block: 128 threads = 4 waves; block tile 128x128x32; wave tile 32x128.
// A tile is staged into LDS via the Tensor Data Mover (wave 0 issues the DMA,
// s_wait_tensorcnt + workgroup barrier publishes it); B tile via vector
// loads + ds stores. 16 v_wmma per K-step.
// ---------------------------------------------------------------------------
__global__ __launch_bounds__(128)
void gemm_f16_wmma(const f16* __restrict__ A, const f16* __restrict__ Bm,
                   const float* __restrict__ bias, const float* __restrict__ resid,
                   float* __restrict__ C,
                   int M, int N, int K, int lda, int ldb, int ldc,
                   int blayout, int epi)
{
  __shared__ __align__(16) f16 As[128][32];
  __shared__ __align__(16) f16 Bs[128][32];

  const int tid  = threadIdx.x;
  const int wv   = tid >> 5;        // wave id 0..3
  const int lane = tid & 31;
  const int l16  = lane & 15;
  const int lhi  = lane >> 4;       // 0 or 1
  const int m0   = blockIdx.y * 128;
  const int n0   = blockIdx.x * 128;

  v8f acc[2][8];
  for (int a = 0; a < 2; ++a)
    for (int b = 0; b < 8; ++b) acc[a][b] = zero_v8f();

#ifdef HAVE_TDM
  // ---- Tensor DMA descriptor for the A tile (2D, f16, tile 32(K) x 128(M)) ----
  const unsigned lds_base = (unsigned)(size_t)(&As[0][0]);
  const unsigned long long gbase =
      (unsigned long long)(size_t)A + ((size_t)m0 * (size_t)lda) * 2ull;
  i32x8 g1;
  g1[0] = 0x00010000;                                     // data_size = 2 bytes
  g1[1] = (int)(((unsigned)K & 0xFFFFu) << 16);           // tensor_dim0[15:0]
  g1[2] = (int)(((unsigned)K >> 16) | (((unsigned)M & 0xFFFFu) << 16));
  g1[3] = (int)(((unsigned)M >> 16) | (32u << 16));       // tile_dim0 = 32
  g1[4] = 128;                                            // tile_dim1 = 128
  g1[5] = lda;                                            // tensor_dim0_stride
  g1[6] = 0;
  g1[7] = 0;
  i32x4 gz; gz[0] = 0; gz[1] = 0; gz[2] = 0; gz[3] = 0;
  i32x8 gz8;
  for (int zz = 0; zz < 8; ++zz) gz8[zz] = 0;
#endif

  for (int k0 = 0; k0 < K; k0 += 32) {
#ifdef HAVE_TDM
    // ---- stage A tile (128x32) via TDM: one DMA per workgroup ----
    if (wv == 0) {
      unsigned long long ga = gbase + (unsigned long long)((unsigned)k0 * 2u);
      u32x4 g0;
      g0[0] = 1u;                                          // count=1, load
      g0[1] = lds_base;                                    // LDS byte address
      g0[2] = (unsigned)ga;                                // global addr lo
      g0[3] = (unsigned)((ga >> 32) & 0x01FFFFFFu) | 0x80000000u;  // hi | type=2
      __builtin_amdgcn_tensor_load_to_lds(g0, g1, gz, gz, gz8, 0);
    }
#else
    // ---- fallback: cooperative A tile load (4 x 16B chunks / thread) ----
#pragma unroll
    for (int i = 0; i < 4; ++i) {
      int c   = tid + i * 128;
      int row = c >> 2;
      int kc  = (c & 3) << 3;
      int gm  = m0 + row;
      v8h vv  = zero_v8h();
      if (gm < M) vv = *(const v8h*)(A + (size_t)gm * lda + (k0 + kc));
      *(v8h*)(&As[row][kc]) = vv;
    }
#endif
    // ---- stage B tile (128x32, stored [n][k]) ----
    if (blayout) {
#pragma unroll
      for (int i = 0; i < 4; ++i) {
        int c   = tid + i * 128;
        int row = c >> 2;
        int kc  = (c & 3) << 3;
        int gn  = n0 + row;
        v8h vv  = zero_v8h();
        if (gn < N) {
          const f16* p = Bm + (size_t)gn * ldb + (k0 + kc);
          vv = *(const v8h*)p;
          if (k0 + 32 < K) __builtin_prefetch(p + 32, 0, 1);
        }
        *(v8h*)(&Bs[row][kc]) = vv;
      }
    } else {
      for (int c = tid; c < 128 * 32; c += 128) {
        int n  = c >> 5;
        int kk = c & 31;
        int gn = n0 + n;
        f16 val = (f16)0.0f;
        if (gn < N) val = Bm[(size_t)(k0 + kk) * ldb + gn];
        Bs[n][kk] = val;
      }
    }
#ifdef HAVE_TDM
    if (wv == 0) __builtin_amdgcn_s_wait_tensorcnt(0);     // DMA landed in LDS
#endif
    __syncthreads();

    // ---- wave computes its 32x128 sub-tile: 2x8 WMMAs ----
#pragma unroll
    for (int mi = 0; mi < 2; ++mi) {
      int arow = (wv << 5) + (mi << 4) + l16;
      int ak   = lhi << 3;  // lanes 0-15: K kb..kb+7 & kb+16..kb+23
      v8h a0 = *(const v8h*)(&As[arow][ak]);
      v8h a1 = *(const v8h*)(&As[arow][ak + 16]);
      v16h af = __builtin_shufflevector(a0, a1, 0,1,2,3,4,5,6,7,8,9,10,11,12,13,14,15);
#pragma unroll
      for (int ni = 0; ni < 8; ++ni) {
        int brow = (ni << 4) + l16;
        int bk2  = lhi << 3;  // lanes 0-15: K 0..7 & 16..23, lanes 16-31: K 8..15 & 24..31
        v8h b0 = *(const v8h*)(&Bs[brow][bk2]);
        v8h b1 = *(const v8h*)(&Bs[brow][bk2 + 16]);
        v16h bfr = __builtin_shufflevector(b0, b1, 0,1,2,3,4,5,6,7,8,9,10,11,12,13,14,15);
        acc[mi][ni] = __builtin_amdgcn_wmma_f32_16x16x32_f16(
            false, af, false, bfr, (short)0, acc[mi][ni], false, false);
      }
    }
    __syncthreads();
  }

  // ---- epilogue: C/D layout -> VGPR j holds M = base + j, N = l16 ----
#pragma unroll
  for (int mi = 0; mi < 2; ++mi) {
    int mb = m0 + (wv << 5) + (mi << 4) + (lhi << 3);
#pragma unroll
    for (int ni = 0; ni < 8; ++ni) {
      int n = n0 + (ni << 4) + l16;
      if (n >= N) continue;
      float bsv = (epi & 1) ? bias[n] : 0.0f;
#pragma unroll
      for (int j = 0; j < 8; ++j) {
        int m = mb + j;
        if (m >= M) continue;
        float r = acc[mi][ni][j] + bsv;
        if (epi & 4) r += resid[(size_t)m * ldc + n];
        if (epi & 2) r = gelu_f(r);
        C[(size_t)m * ldc + n] = r;
      }
    }
  }
}

// ---------------------------------------------------------------------------
// LayerNorm over D=512 per row (block=256, 2 elems/thread), optional GELU.
// ---------------------------------------------------------------------------
__global__ __launch_bounds__(256)
void k_ln(const float* __restrict__ x, const float* __restrict__ g, const float* __restrict__ b,
          float* __restrict__ y, int do_gelu)
{
  __shared__ float red[256];
  int row = blockIdx.x, tid = threadIdx.x;
  const float* xr = x + ((size_t)row << 9);
  float v0 = xr[tid], v1 = xr[tid + 256];
  red[tid] = v0 + v1; __syncthreads();
  for (int o2 = 128; o2 > 0; o2 >>= 1) { if (tid < o2) red[tid] += red[tid + o2]; __syncthreads(); }
  float mean = red[0] * (1.0f / 512.0f);
  __syncthreads();
  float d0 = v0 - mean, d1 = v1 - mean;
  red[tid] = d0 * d0 + d1 * d1; __syncthreads();
  for (int o2 = 128; o2 > 0; o2 >>= 1) { if (tid < o2) red[tid] += red[tid + o2]; __syncthreads(); }
  float rs = rsqrtf(red[0] * (1.0f / 512.0f) + 1e-5f);
  float r0 = d0 * rs * g[tid] + b[tid];
  float r1 = d1 * rs * g[tid + 256] + b[tid + 256];
  if (do_gelu) { r0 = gelu_f(r0); r1 = gelu_f(r1); }
  float* yr = y + ((size_t)row << 9);
  yr[tid] = r0; yr[tid + 256] = r1;
}

// ---------------------------------------------------------------------------
// Elementwise helpers (grid-stride)
// ---------------------------------------------------------------------------
__global__ void k_cvt16(const float* __restrict__ a, f16* __restrict__ o, size_t n) {
  for (size_t i = blockIdx.x * (size_t)blockDim.x + threadIdx.x; i < n; i += (size_t)gridDim.x * blockDim.x)
    o[i] = (f16)a[i];
}
__global__ void k_zero(float* __restrict__ p, size_t n) {
  for (size_t i = blockIdx.x * (size_t)blockDim.x + threadIdx.x; i < n; i += (size_t)gridDim.x * blockDim.x)
    p[i] = 0.0f;
}
__global__ void k_gelu_ip(float* __restrict__ p, size_t n) {
  for (size_t i = blockIdx.x * (size_t)blockDim.x + threadIdx.x; i < n; i += (size_t)gridDim.x * blockDim.x)
    p[i] = gelu_f(p[i]);
}
__global__ void k_acc_weighted(float* __restrict__ acc, const float* __restrict__ r,
                               const float* __restrict__ sw, int which, size_t n) {
  for (size_t i = blockIdx.x * (size_t)blockDim.x + threadIdx.x; i < n; i += (size_t)gridDim.x * blockDim.x) {
    int b = (int)(i >> 18);           // T*D = 2^18 elems per batch
    acc[i] += sw[b * 3 + which] * r[i];
  }
}
__global__ void k_final_add(const float* __restrict__ a, const float* __restrict__ c,
                            float* __restrict__ o, size_t n) {
  for (size_t i = blockIdx.x * (size_t)blockDim.x + threadIdx.x; i < n; i += (size_t)gridDim.x * blockDim.x)
    o[i] = a[i] + c[i];
}

// ---------------------------------------------------------------------------
// STFT path: framing (reflect pad + hann window), DFT basis, magnitude reductions
// ---------------------------------------------------------------------------
__global__ void k_build_dft(f16* __restrict__ bd) {
  const size_t n = (size_t)DFTN * 512;
  for (size_t idx = blockIdx.x * (size_t)blockDim.x + threadIdx.x; idx < n; idx += (size_t)gridDim.x * blockDim.x) {
    int r = (int)(idx >> 9);
    int t = (int)(idx & 511);
    int f = r >> 1;
    float ang = 6.28318530718f * (float)((f * t) & 511) * (1.0f / 512.0f);
    float v = (r & 1) ? __sinf(ang) : __cosf(ang);
    bd[idx] = (f16)v;
  }
}
__global__ void k_frames(const float* __restrict__ x, f16* __restrict__ Fc, int r0) {
  const size_t n = (size_t)CHUNK * 512;
  for (size_t idx = blockIdx.x * (size_t)blockDim.x + threadIdx.x; idx < n; idx += (size_t)gridDim.x * blockDim.x) {
    int lr = (int)(idx >> 9);
    int tt = (int)(idx & 511);
    int r = r0 + lr;
    int b = r / (NFRM * 512);
    int rem = r - b * (NFRM * 512);
    int fr = rem >> 9;
    int ch = rem & 511;
    float wv = 0.0f;
    if (tt >= 232 && tt < 280) {           // hann(48) centered in n_fft=512
      float ph = (float)(tt - 232) * (6.28318530718f / 48.0f);
      wv = 0.5f - 0.5f * __cosf(ph);
    }
    int src = fr * 48 + tt - 256;          // center pad n_fft/2, reflect
    if (src < 0) src = -src;
    if (src > 511) src = 1022 - src;
    float val = x[(((size_t)b << 9) + src) * 512 + ch];
    Fc[idx] = (f16)(val * wv);
  }
}
__global__ void k_magacc(const float* __restrict__ G, float* __restrict__ freq) {
  const size_t n = (size_t)CHUNK * NFREQ;
  for (size_t idx = blockIdx.x * (size_t)blockDim.x + threadIdx.x; idx < n; idx += (size_t)gridDim.x * blockDim.x) {
    int r = (int)(idx / NFREQ);
    int f = (int)(idx - (size_t)r * NFREQ);
    float re = G[(size_t)r * DFTN + 2 * f];
    float im = G[(size_t)r * DFTN + 2 * f + 1];
    atomicAdd(&freq[f], sqrtf(re * re + im * im));
  }
}
__global__ void k_topk(float* __restrict__ freq, int* __restrict__ tp) {
  if (blockIdx.x == 0 && threadIdx.x == 0) {
    freq[0] = 0.0f;
    bool used[NFREQ];
    for (int j = 0; j < NFREQ; ++j) used[j] = false;
    for (int kk = 0; kk < 3; ++kk) {
      int best = 1; float bv = -1.0f;
      for (int j = 0; j < NFREQ; ++j)
        if (!used[j] && freq[j] > bv) { bv = freq[j]; best = j; }
      used[best] = true;
      tp[kk] = best;
      int p = best > 0 ? 512 / best : 512;
      if (p < 1) p = 1; if (p > 512) p = 512;
      tp[3 + kk] = p;
    }
  }
}
__global__ void k_swacc(const float* __restrict__ G, const int* __restrict__ tp,
                        float* __restrict__ swsum, int r0) {
  const size_t n = (size_t)CHUNK * 3;
  for (size_t idx = blockIdx.x * (size_t)blockDim.x + threadIdx.x; idx < n; idx += (size_t)gridDim.x * blockDim.x) {
    int lr = (int)(idx / 3);
    int kk = (int)(idx - (size_t)lr * 3);
    int r = r0 + lr;
    int b = r / (NFRM * 512);
    int f = tp[kk];
    float re = G[(size_t)lr * DFTN + 2 * f];
    float im = G[(size_t)lr * DFTN + 2 * f + 1];
    atomicAdd(&swsum[b * 3 + kk], sqrtf(re * re + im * im));
  }
}
__global__ void k_swsoftmax(const float* __restrict__ swsum, float* __restrict__ sw) {
  int b = threadIdx.x;
  if (b < NB) {
    float inv = 1.0f / (float)(NFRM * 512);
    float s0 = swsum[b * 3 + 0] * inv;
    float s1 = swsum[b * 3 + 1] * inv;
    float s2 = swsum[b * 3 + 2] * inv;
    float mx = fmaxf(s0, fmaxf(s1, s2));
    float e0 = __expf(s0 - mx), e1 = __expf(s1 - mx), e2 = __expf(s2 - mx);
    float s = e0 + e1 + e2;
    sw[b * 3 + 0] = e0 / s; sw[b * 3 + 1] = e1 / s; sw[b * 3 + 2] = e2 / s;
  }
}

// ---------------------------------------------------------------------------
// Graph block helpers
// ---------------------------------------------------------------------------
__global__ __launch_bounds__(1024)
void k_adj(const float* __restrict__ nv1, const float* __restrict__ nv2, float* __restrict__ am) {
  __shared__ float e[32][32];
  __shared__ float aj[32][32];
  int v = threadIdx.y, w2 = threadIdx.x;
  float s = 0.0f;
  for (int d = 0; d < 10; ++d) s += nv1[v * 10 + d] * nv2[d * 32 + w2];
  s = fmaxf(s, 0.0f);
  e[v][w2] = s; __syncthreads();
  float mx = -1e30f; for (int j = 0; j < 32; ++j) mx = fmaxf(mx, e[v][j]);
  float sm = 0.0f;   for (int j = 0; j < 32; ++j) sm += __expf(e[v][j] - mx);
  float adjv = __expf(s - mx) / sm + ((v == w2) ? 1.0f : 0.0f);
  aj[v][w2] = adjv; __syncthreads();
  float rs = 0.0f;   for (int j = 0; j < 32; ++j) rs += aj[v][j];
  am[v * 32 + w2] = adjv / rs;
}
// bake 481-tap sliding conv into a banded [1024 x 512] f16 weight (NT layout)
__global__ void k_build_wg(const float* __restrict__ sw_, f16* __restrict__ wg) {
  const size_t n = (size_t)GCH * 512;
  for (size_t idx = blockIdx.x * (size_t)blockDim.x + threadIdx.x; idx < n; idx += (size_t)gridDim.x * blockDim.x) {
    int nn = (int)(idx >> 9);
    int d  = (int)(idx & 511);
    int o = nn >> 5, v = nn & 31;
    int j = d - v;
    float val = (j >= 0 && j < 481) ? sw_[o * 481 + j] : 0.0f;
    wg[idx] = (f16)val;
  }
}
__global__ void k_bexp(const float* __restrict__ sb, float* __restrict__ bn) {
  int n = blockIdx.x * blockDim.x + threadIdx.x;
  if (n < GCH) bn[n] = sb[n >> 5];
}
// h' = alpha*x0 + (1-alpha) * (A h) along the 32 node dim
__global__ void k_prop(const float* __restrict__ hin, const float* __restrict__ h0,
                       const float* __restrict__ am, float* __restrict__ hout) {
  const size_t n = (size_t)NROWS * GCH;
  for (size_t idx = blockIdx.x * (size_t)blockDim.x + threadIdx.x; idx < n; idx += (size_t)gridDim.x * blockDim.x) {
    size_t row = idx >> 10;
    int c = (int)(idx & 1023);
    int o = c >> 5, v = c & 31;
    const float* hr = hin + (row << 10) + (o << 5);
    const float* ar = am + v * 32;
    float s = 0.0f;
    for (int w = 0; w < 32; ++w) s += ar[w] * hr[w];
    hout[idx] = 0.05f * h0[idx] + 0.95f * s;
  }
}
// ho[row,(sc,v)] (+)= sum_c mw[sc, depth*32+c] * h[row,(c,v)]   (init adds mb[sc])
__global__ void k_mix(const float* __restrict__ h, const float* __restrict__ mw,
                      const float* __restrict__ mb, float* __restrict__ ho, int depth, int init) {
  const size_t n = (size_t)NROWS * GCH;
  for (size_t idx = blockIdx.x * (size_t)blockDim.x + threadIdx.x; idx < n; idx += (size_t)gridDim.x * blockDim.x) {
    size_t row = idx >> 10;
    int c = (int)(idx & 1023);
    int sc = c >> 5, v = c & 31;
    const float* hr = h + (row << 10) + v;          // stride 32 over c
    const float* m = mw + sc * 96 + depth * 32;
    float s = init ? mb[sc] : ho[idx];
    for (int cc = 0; cc < 32; ++cc) s += m[cc] * hr[cc << 5];
    ho[idx] = s;
  }
}
// PA[(b,n), c*512+s] = ho[(b,s),(c,n)]  (f16 pack for end-einsum GEMM)
__global__ void k_packpa(const float* __restrict__ ho, f16* __restrict__ pa) {
  const size_t n = (size_t)GCH * 16384;
  for (size_t idx = blockIdx.x * (size_t)blockDim.x + threadIdx.x; idx < n; idx += (size_t)gridDim.x * blockDim.x) {
    size_t rown = idx >> 14;
    int kk = (int)(idx & 16383);
    int sc = kk >> 9, tm = kk & 511;
    int b = (int)(rown >> 5), nn = (int)(rown & 31);
    pa[idx] = (f16)ho[((((size_t)b << 9) + tm) << 10) + (sc << 5) + nn];
  }
}
// ob[(b,t), n] = oute[(b,n), t]  (f16 pack for linear GEMM)
__global__ void k_packlin(const float* __restrict__ oute, f16* __restrict__ ob) {
  const size_t n = (size_t)NROWS * 32;
  for (size_t idx = blockIdx.x * (size_t)blockDim.x + threadIdx.x; idx < n; idx += (size_t)gridDim.x * blockDim.x) {
    size_t row = idx >> 5;
    int nn = (int)(idx & 31);
    int b = (int)(row >> 9), t = (int)(row & 511);
    ob[idx] = (f16)oute[((((size_t)b << 5) + nn) << 9) + t];
  }
}

// ---------------------------------------------------------------------------
// Segmented attention, streaming softmax. One wave per (b,t,h) query row.
// Segment length read from device (data-dependent period); fixed max grid.
// ---------------------------------------------------------------------------
__global__ __launch_bounds__(32)
void k_attn(const float* __restrict__ q, const float* __restrict__ k,
            const float* __restrict__ v, float* __restrict__ o,
            const int* __restrict__ tp, int which)
{
  int scale = tp[3 + which];
  if (scale < 1) scale = 1; if (scale > 512) scale = 512;
  int gid = blockIdx.x;
  int h  = gid & 7;
  int bt = gid >> 3;
  int t  = bt & 511;
  int b  = bt >> 9;
  int lane = threadIdx.x;
  const float* qr = q + ((size_t)bt << 9) + (h << 6);
  float q0 = qr[lane] * 0.125f, q1 = qr[lane + 32] * 0.125f;  // 1/sqrt(64)
  int s0 = (t / scale) * scale;
  int s1 = s0 + scale; if (s1 > 512) s1 = 512;
  float mx = -1e30f, den = 0.0f, a0 = 0.0f, a1 = 0.0f;
  for (int t2 = s0; t2 < s1; ++t2) {
    size_t rr = (((size_t)b << 9) + t2) << 9;
    const float* kr = k + rr + (h << 6);
    float p = q0 * kr[lane] + q1 * kr[lane + 32];
#pragma unroll
    for (int off2 = 16; off2 > 0; off2 >>= 1) p += __shfl_xor(p, off2, 32);
    float nmx = fmaxf(mx, p);
    float cf = __expf(mx - nmx);
    float wf = __expf(p - nmx);
    const float* vr = v + rr + (h << 6);
    a0 = a0 * cf + wf * vr[lane];
    a1 = a1 * cf + wf * vr[lane + 32];
    den = den * cf + wf;
    mx = nmx;
  }
  float inv = 1.0f / den;
  float* orow = o + ((size_t)bt << 9) + (h << 6);
  orow[lane] = a0 * inv; orow[lane + 32] = a1 * inv;
}

// ---------------------------------------------------------------------------
// Host orchestration
// ---------------------------------------------------------------------------
static inline void gemm(hipStream_t s, const f16* A, const f16* Bm, const float* bias,
                        const float* resid, float* C, int M, int N, int K,
                        int lda, int ldb, int ldc, int blayout, int epi) {
  dim3 g((N + 127) / 128, (M + 127) / 128), b(128);
  gemm_f16_wmma<<<g, b, 0, s>>>(A, Bm, bias, resid, C, M, N, K, lda, ldb, ldc, blayout, epi);
}

extern "C" void kernel_launch(void* const* d_in, const int* in_sizes, int n_in,
                              void* d_out, int out_size, void* d_ws, size_t ws_size,
                              hipStream_t stream)
{
  (void)in_sizes; (void)n_in; (void)out_size; (void)ws_size;
  const float* X    = (const float*)d_in[0];
  const float* NV1  = (const float*)d_in[1];
  const float* NV2  = (const float*)d_in[2];
  const float* SWC  = (const float*)d_in[3];   // start_w
  const float* SBC  = (const float*)d_in[4];   // start_b
  const float* MW   = (const float*)d_in[5];
  const float* MB   = (const float*)d_in[6];
  const float* EW   = (const float*)d_in[7];
  const float* EB   = (const float*)d_in[8];
  const float* LWp  = (const float*)d_in[9];
  const float* LBp  = (const float*)d_in[10];
  const float* GG   = (const float*)d_in[11];
  const float* GBb  = (const float*)d_in[12];
  const float* WQ   = (const float*)d_in[13];
  const float* BQ   = (const float*)d_in[14];
  const float* WK   = (const float*)d_in[15];
  const float* BKb  = (const float*)d_in[16];
  const float* WV   = (const float*)d_in[17];
  const float* BV   = (const float*)d_in[18];
  const float* WO   = (const float*)d_in[19];
  const float* BO   = (const float*)d_in[20];
  const float* F1W  = (const float*)d_in[21];
  const float* F1B  = (const float*)d_in[22];
  const float* F2W  = (const float*)d_in[23];
  const float* F2B  = (const float*)d_in[24];
  const float* A1G  = (const float*)d_in[25];
  const float* A1B  = (const float*)d_in[26];
  const float* A2G  = (const float*)d_in[27];
  const float* A2B  = (const float*)d_in[28];
  const float* NG   = (const float*)d_in[29];
  const float* NBi  = (const float*)d_in[30];
  float* OUT = (float*)d_out;

  const size_t AF = (size_t)NROWS * ND;        // activation elems

  char* wsp = (char*)d_ws;
  size_t off = 0;
  auto alloc = [&](size_t bytes) -> char* {
    char* p = wsp + off;
    off += (bytes + 255) & ~((size_t)255);
    return p;
  };
  float* cur    = (float*)alloc(AF * 4);
  float* t1     = (float*)alloc(AF * 4);
  float* t2     = (float*)alloc(AF * 4);
  float* x1     = (float*)alloc(AF * 4);
  float* attn   = (float*)alloc(AF * 4);
  float* qb     = (float*)alloc(AF * 4);
  float* kb     = (float*)alloc(AF * 4);
  float* vb     = (float*)alloc(AF * 4);
  float* outacc = (float*)alloc(AF * 4);
  float* resb   = (float*)alloc(AF * 4);
  f16*   curh   = (f16*)alloc(AF * 2);
  f16*   h16    = (f16*)alloc(AF * 2);
  float* hA     = (float*)alloc((size_t)NROWS * GCH * 4);
  float* hB     = (float*)alloc((size_t)NROWS * GCH * 4);
  float* hC     = (float*)alloc((size_t)NROWS * GCH * 4);
  float* ho     = (float*)alloc((size_t)NROWS * GCH * 4);
  f16*   PA     = (f16*)alloc((size_t)GCH * 16384 * 2);
  float* oute   = (float*)alloc((size_t)GCH * 512 * 4);
  f16*   ob16   = (f16*)alloc((size_t)NROWS * 32 * 2);
  f16*   wg16   = (f16*)alloc((size_t)GCH * 512 * 2);
  float* biasN  = (float*)alloc(GCH * 4);
  float* amat   = (float*)alloc(32 * 32 * 4);
  f16*   ew16   = (f16*)alloc((size_t)512 * 16384 * 2);
  f16*   wq16   = (f16*)alloc((size_t)512 * 512 * 2);
  f16*   wk16   = (f16*)alloc((size_t)512 * 512 * 2);
  f16*   wv16   = (f16*)alloc((size_t)512 * 512 * 2);
  f16*   wo16   = (f16*)alloc((size_t)512 * 512 * 2);
  f16*   f1_16  = (f16*)alloc((size_t)512 * 512 * 2);
  f16*   f2_16  = (f16*)alloc((size_t)512 * 512 * 2);
  f16*   lw16   = (f16*)alloc((size_t)32 * 512 * 2);
  f16*   bd16   = (f16*)alloc((size_t)DFTN * 512 * 2);
  f16*   Fc     = (f16*)alloc((size_t)CHUNK * 512 * 2);
  float* Gc     = (float*)alloc((size_t)CHUNK * DFTN * 4);
  float* freq   = (float*)alloc(NFREQ * 4);
  int*   tp     = (int*)alloc(8 * 4);
  float* swsum  = (float*)alloc(NB * 3 * 4);
  float* swb    = (float*)alloc(NB * 3 * 4);

  dim3 eg(2048), eb(256);

  // cur = x
  (void)hipMemcpyAsync(cur, X, AF * 4, hipMemcpyDeviceToDevice, stream);

  // weight conversions (iteration independent)
  k_cvt16<<<eg, eb, 0, stream>>>(WQ,  wq16, (size_t)512 * 512);
  k_cvt16<<<eg, eb, 0, stream>>>(WK,  wk16, (size_t)512 * 512);
  k_cvt16<<<eg, eb, 0, stream>>>(WV,  wv16, (size_t)512 * 512);
  k_cvt16<<<eg, eb, 0, stream>>>(WO,  wo16, (size_t)512 * 512);
  k_cvt16<<<eg, eb, 0, stream>>>(F1W, f1_16, (size_t)512 * 512);
  k_cvt16<<<eg, eb, 0, stream>>>(F2W, f2_16, (size_t)512 * 512);

  // ---------------- STFT-as-GEMM: period / weight estimation ----------------
  k_zero<<<eg, eb, 0, stream>>>(freq, NFREQ);
  k_zero<<<eg, eb, 0, stream>>>(swsum, NB * 3);
  k_build_dft<<<eg, eb, 0, stream>>>(bd16);
  const int NCHUNK = (NB * NFRM * 512) / CHUNK;   // 22
  for (int c = 0; c < NCHUNK; ++c) {              // pass 1: freq spectrum sums
    k_frames<<<eg, eb, 0, stream>>>(X, Fc, c * CHUNK);
    gemm(stream, Fc, bd16, nullptr, nullptr, Gc, CHUNK, DFTN, 512, 512, 512, DFTN, 1, 0);
    k_magacc<<<eg, eb, 0, stream>>>(Gc, freq);
  }
  k_topk<<<1, 1, 0, stream>>>(freq, tp);
  for (int c = 0; c < NCHUNK; ++c) {              // pass 2: per-batch weights
    k_frames<<<eg, eb, 0, stream>>>(X, Fc, c * CHUNK);
    gemm(stream, Fc, bd16, nullptr, nullptr, Gc, CHUNK, DFTN, 512, 512, 512, DFTN, 1, 0);
    k_swacc<<<eg, eb, 0, stream>>>(Gc, tp, swsum, c * CHUNK);
  }
  k_swsoftmax<<<1, 32, 0, stream>>>(swsum, swb);

  k_zero<<<eg, eb, 0, stream>>>(outacc, AF);

  // ---------------- TOP_K pipeline ----------------
  for (int i = 0; i < 3; ++i) {
    // ---- graph block ----
    k_adj<<<1, dim3(32, 32), 0, stream>>>(NV1 + (size_t)i * 320, NV2 + (size_t)i * 320, amat);
    k_build_wg<<<eg, eb, 0, stream>>>(SWC + (size_t)i * 32 * 481, wg16);
    k_bexp<<<1, 1024, 0, stream>>>(SBC + (size_t)i * 32, biasN);
    k_cvt16<<<eg, eb, 0, stream>>>(cur, curh, AF);
    gemm(stream, curh, wg16, biasN, nullptr, hA, NROWS, GCH, 512, 512, 512, GCH, 1, 1);
    k_mix<<<eg, eb, 0, stream>>>(hA, MW + (size_t)i * 32 * 96, MB + (size_t)i * 32, ho, 0, 1);
    k_prop<<<eg, eb, 0, stream>>>(hA, hA, amat, hB);
    k_mix<<<eg, eb, 0, stream>>>(hB, MW + (size_t)i * 32 * 96, MB + (size_t)i * 32, ho, 1, 0);
    k_prop<<<eg, eb, 0, stream>>>(hB, hA, amat, hC);
    k_mix<<<eg, eb, 0, stream>>>(hC, MW + (size_t)i * 32 * 96, MB + (size_t)i * 32, ho, 2, 0);
    k_gelu_ip<<<eg, eb, 0, stream>>>(ho, (size_t)NROWS * GCH);
    k_packpa<<<eg, eb, 0, stream>>>(ho, PA);
    k_cvt16<<<eg, eb, 0, stream>>>(EW + (size_t)i * 512 * 32 * 512, ew16, (size_t)512 * 16384);
    gemm(stream, PA, ew16, EB + (size_t)i * 512, nullptr, oute,
         GCH, 512, 16384, 16384, 16384, 512, 1, 1);
    k_packlin<<<eg, eb, 0, stream>>>(oute, ob16);
    k_cvt16<<<eg, eb, 0, stream>>>(LWp + (size_t)i * 32 * 512, lw16, (size_t)32 * 512);
    gemm(stream, ob16, lw16, LBp + (size_t)i * 512, cur, t1,
         NROWS, 512, 32, 32, 512, 512, 0, 1 | 4);
    k_ln<<<dim3(NROWS), dim3(256), 0, stream>>>(t1, GG + (size_t)i * 512, GBb + (size_t)i * 512, cur, 0);

    // ---- attention block (period-segmented) ----
    k_cvt16<<<eg, eb, 0, stream>>>(cur, curh, AF);
    gemm(stream, curh, wq16, BQ,  nullptr, qb, NROWS, 512, 512, 512, 512, 512, 1, 1);
    gemm(stream, curh, wk16, BKb, nullptr, kb, NROWS, 512, 512, 512, 512, 512, 1, 1);
    gemm(stream, curh, wv16, BV,  nullptr, vb, NROWS, 512, 512, 512, 512, 512, 1, 1);
    k_attn<<<dim3(NROWS * NH), dim3(32), 0, stream>>>(qb, kb, vb, attn, tp, i);
    k_cvt16<<<eg, eb, 0, stream>>>(attn, h16, AF);
    gemm(stream, h16, wo16, BO, cur, t1, NROWS, 512, 512, 512, 512, 512, 1, 1 | 4);
    k_ln<<<dim3(NROWS), dim3(256), 0, stream>>>(t1, A1G, A1B, x1, 0);
    k_cvt16<<<eg, eb, 0, stream>>>(x1, h16, AF);
    gemm(stream, h16, f1_16, F1B, nullptr, t2, NROWS, 512, 512, 512, 512, 512, 1, 1 | 2);
    k_cvt16<<<eg, eb, 0, stream>>>(t2, h16, AF);
    gemm(stream, h16, f2_16, F2B, x1, t1, NROWS, 512, 512, 512, 512, 512, 1, 1 | 4);
    k_ln<<<dim3(NROWS), dim3(256), 0, stream>>>(t1, A2G, A2B, t2, 0);
    k_ln<<<dim3(NROWS), dim3(256), 0, stream>>>(t2, NG, NBi, resb, 1);  // gelu(LN(...))
    k_acc_weighted<<<eg, eb, 0, stream>>>(outacc, resb, swb, i, AF);
  }

  k_final_add<<<eg, eb, 0, stream>>>(outacc, cur, OUT, AF);
}
